// SparseLocalAttention_28475633172717
// MI455X (gfx1250) — compile-verified, matrix operand order checked by reading, not run on hardware
//
#include <hip/hip_runtime.h>
#include <math.h>

// ---------------------------------------------------------------------------
// SparseLocalAttention for MI455X (gfx1250, wave32)
//   Pass 1: fused Q/K/V projection as bf16 WMMA GEMM (f32 accumulate)
//   Pass 2: 7-wide local attention, one wave32 per sequence position
// ---------------------------------------------------------------------------

typedef __attribute__((ext_vector_type(16))) __bf16 bf16x16;
typedef __attribute__((ext_vector_type(8)))  float  f32x8;

#define DDIM  256
#define NSEQ  4096
#define BATCH 8
#define MROWS (BATCH * NSEQ)   // 32768
#define WIN   3
#define NW    7                // 2*WIN + 1

__device__ __forceinline__ unsigned short f32_to_bf16_rne(float f) {
    unsigned int u = __float_as_uint(f);
    unsigned int r = 0x7FFFu + ((u >> 16) & 1u);
    return (unsigned short)((u + r) >> 16);
}

union BFrag {
    bf16x16        v;
    unsigned short u[16];
    uint4          q[2];
};

// ---------------------------------------------------------------------------
// Pass 1: out[m, n] = sum_k x[m,k] * W[n,k] + bias[n]   (W row-major, D x D)
// gridDim.x = MROWS/128, gridDim.y = 3 (selects Q / K / V)
// 8 waves per block, each wave computes a 16-row M tile across all 256 cols.
// ---------------------------------------------------------------------------
__global__ void __launch_bounds__(256)
qkv_proj_kernel(const float* __restrict__ x,
                const float* __restrict__ Wq, const float* __restrict__ bq,
                const float* __restrict__ Wk, const float* __restrict__ bk,
                const float* __restrict__ Wv, const float* __restrict__ bv,
                float* __restrict__ Q, float* __restrict__ Kp, float* __restrict__ Vp)
{
    __shared__ unsigned short sW[64 * DDIM];   // 32 KB: 64 output cols of W, bf16

    const float* W;  const float* bias;  float* out;
    if (blockIdx.y == 0)      { W = Wq; bias = bq; out = Q;  }
    else if (blockIdx.y == 1) { W = Wk; bias = bk; out = Kp; }
    else                      { W = Wv; bias = bv; out = Vp; }

    const int tid      = threadIdx.x;
    const int lane     = tid & 31;
    const int wave     = tid >> 5;
    const int laneHalf = lane >> 4;     // 0: lanes 0-15, 1: lanes 16-31
    const int m16      = lane & 15;

    const int rowStart = blockIdx.x * 128 + wave * 16;

    // ---- Preload all 8 A fragments (16x32 bf16 each) for this wave's rows.
    // ISA A layout: lane L%16 holds row M=L%16; lanes 0-15 carry K {0..7,16..23},
    // lanes 16-31 carry K {8..15,24..31} within each 32-wide K chunk.
    const float* xrow = x + (size_t)(rowStart + m16) * DDIM;
    bf16x16 af[8];
    #pragma unroll
    for (int kc = 0; kc < 8; ++kc) {
        const int kb = kc * 32 + laneHalf * 8;
        BFrag a;
        #pragma unroll
        for (int j = 0; j < 8; ++j) a.u[j]     = f32_to_bf16_rne(xrow[kb + j]);
        #pragma unroll
        for (int j = 0; j < 8; ++j) a.u[8 + j] = f32_to_bf16_rne(xrow[kb + 16 + j]);
        af[kc] = a.v;
    }

    // ---- 4 stages: stage 64 output-columns of W into LDS, do 4 N tiles each.
    for (int s = 0; s < 4; ++s) {
        const int colBase = s * 64;
        __syncthreads();
        // cooperative f32 -> bf16 staging (float4 in, ushort4 out)
        #pragma unroll 4
        for (int i = tid; i < (64 * DDIM) / 4; i += 256) {
            const int e = i * 4;
            const float4 f = *(const float4*)&W[(size_t)colBase * DDIM + e];
            ushort4 h;
            h.x = f32_to_bf16_rne(f.x);
            h.y = f32_to_bf16_rne(f.y);
            h.z = f32_to_bf16_rne(f.z);
            h.w = f32_to_bf16_rne(f.w);
            *(ushort4*)&sW[e] = h;
        }
        __syncthreads();

        #pragma unroll
        for (int ntl = 0; ntl < 4; ++ntl) {
            f32x8 acc = {};
            const int colLocal = ntl * 16 + m16;   // B col N = lane%16 within tile
            #pragma unroll
            for (int kc = 0; kc < 8; ++kc) {
                const int kb = kc * 32 + laneHalf * 8;
                BFrag bf;
                bf.q[0] = *(const uint4*)&sW[colLocal * DDIM + kb];        // K kb..kb+7
                bf.q[1] = *(const uint4*)&sW[colLocal * DDIM + kb + 16];   // K kb+16..kb+23
                acc = __builtin_amdgcn_wmma_f32_16x16x32_bf16(
                          false, af[kc], false, bf.v, (short)0, acc, false, false);
            }
            // C layout: lanes 0-15 VGPR r -> M=r; lanes 16-31 VGPR r -> M=8+r.
            const int   col   = colBase + ntl * 16 + m16;
            const float bb    = bias[col];
            const int   mBase = rowStart + laneHalf * 8;
            #pragma unroll
            for (int r = 0; r < 8; ++r)
                out[(size_t)(mBase + r) * DDIM + col] = acc[r] + bb;
        }
    }
}

// ---------------------------------------------------------------------------
// Pass 2: one wave32 per (b, n) row. Each lane owns 8 contiguous f32 of D.
// ---------------------------------------------------------------------------
__global__ void __launch_bounds__(256)
local_attn_kernel(const float* __restrict__ Q, const float* __restrict__ K,
                  const float* __restrict__ V, float* __restrict__ out)
{
    const int lane = threadIdx.x & 31;
    const int row  = (int)((blockIdx.x * blockDim.x + threadIdx.x) >> 5); // 0..MROWS-1
    const int n    = row & (NSEQ - 1);

    const float4* q4 = (const float4*)(Q + (size_t)row * DDIM);
    const float4  q0 = q4[lane * 2];
    const float4  q1 = q4[lane * 2 + 1];

    float sc[NW];
    #pragma unroll
    for (int w = 0; w < NW; ++w) {
        const int pos = n + w - WIN;
        if (pos >= 0 && pos < NSEQ) {
            const float4* k4 = (const float4*)(K + (size_t)(row + w - WIN) * DDIM);
            const float4 k0 = k4[lane * 2];
            const float4 k1 = k4[lane * 2 + 1];
            float p = q0.x * k0.x + q0.y * k0.y + q0.z * k0.z + q0.w * k0.w
                    + q1.x * k1.x + q1.y * k1.y + q1.z * k1.z + q1.w * k1.w;
            #pragma unroll
            for (int off = 16; off >= 1; off >>= 1)
                p += __shfl_xor(p, off, 32);
            sc[w] = p * (1.0f / 16.0f);          // 1/sqrt(256)
        } else {
            sc[w] = -__builtin_inff();
        }
    }

    float mx = sc[0];
    #pragma unroll
    for (int w = 1; w < NW; ++w) mx = fmaxf(mx, sc[w]);
    float pr[NW];
    float sum = 0.0f;
    #pragma unroll
    for (int w = 0; w < NW; ++w) { pr[w] = __expf(sc[w] - mx); sum += pr[w]; }
    const float inv = 1.0f / sum;

    float4 a0 = {0.f, 0.f, 0.f, 0.f};
    float4 a1 = {0.f, 0.f, 0.f, 0.f};
    #pragma unroll
    for (int w = 0; w < NW; ++w) {
        const int pos = n + w - WIN;
        if (pos >= 0 && pos < NSEQ) {
            const float4* v4 = (const float4*)(V + (size_t)(row + w - WIN) * DDIM);
            const float4 v0 = v4[lane * 2];
            const float4 v1 = v4[lane * 2 + 1];
            const float p = pr[w] * inv;
            a0.x += p * v0.x; a0.y += p * v0.y; a0.z += p * v0.z; a0.w += p * v0.w;
            a1.x += p * v1.x; a1.y += p * v1.y; a1.z += p * v1.z; a1.w += p * v1.w;
        }
    }

    float4* o4 = (float4*)(out + (size_t)row * DDIM);
    o4[lane * 2]     = a0;
    o4[lane * 2 + 1] = a1;
}

// ---------------------------------------------------------------------------
extern "C" void kernel_launch(void* const* d_in, const int* in_sizes, int n_in,
                              void* d_out, int out_size, void* d_ws, size_t ws_size,
                              hipStream_t stream) {
    const float* x  = (const float*)d_in[0];
    const float* Wq = (const float*)d_in[1];
    const float* bq = (const float*)d_in[2];
    const float* Wk = (const float*)d_in[3];
    const float* bk = (const float*)d_in[4];
    const float* Wv = (const float*)d_in[5];
    const float* bv = (const float*)d_in[6];
    float* out = (float*)d_out;

    float* Q = (float*)d_ws;
    float* K = Q + (size_t)MROWS * DDIM;
    float* V = K + (size_t)MROWS * DDIM;

    dim3 g1(MROWS / 128, 3);
    qkv_proj_kernel<<<g1, 256, 0, stream>>>(x, Wq, bq, Wk, bk, Wv, bv, Q, K, V);

    local_attn_kernel<<<(MROWS * 32) / 256, 256, 0, stream>>>(Q, K, V, out);
}
